// MaskedMultiHeadSelfAttention_9388798509561
// MI455X (gfx1250) — compile-verified, matrix operand order checked by reading, not run on hardware
//
#include <hip/hip_runtime.h>

#define DM   768
#define NH   12
#define DK   64
#define BATCH 4
#define SEQ  2048
#define NKSTEP (DM / 32)   // 24 K-steps of 32

typedef __attribute__((ext_vector_type(16))) _Float16 v16h;
typedef __attribute__((ext_vector_type(8)))  _Float16 v8h;
typedef __attribute__((ext_vector_type(8)))  float    v8f;

// ---------------------------------------------------------------------------
// 16-bit A/B fragment for v_wmma_*_16x16x32_f16 (wave32):
// element j (0..15) holds K = (j/8)*16 + half*8 + (j%8)  ->  two contiguous
// 16-byte loads at rowbase + kbase + half*8 and +16.
// ---------------------------------------------------------------------------
__device__ __forceinline__ v16h frag_ld(const _Float16* rowbase, int kbase, int half) {
    v8h lo = *(const v8h*)(rowbase + kbase + half * 8);
    v8h hi = *(const v8h*)(rowbase + kbase + 16 + half * 8);
    v16h a;
#pragma unroll
    for (int i = 0; i < 8; ++i) { a[i] = lo[i]; a[i + 8] = hi[i]; }
    return a;
}

__device__ __forceinline__ v8f wmma16(v16h a, v16h b, v8f c) {
    return __builtin_amdgcn_wmma_f32_16x16x32_f16(false, a, false, b, (short)0, c,
                                                  false, false);
}

__device__ __forceinline__ v8f vzero8() {
    v8f z;
#pragma unroll
    for (int i = 0; i < 8; ++i) z[i] = 0.0f;
    return z;
}

__device__ __forceinline__ void cvt_store16(_Float16* dst, const float4* g) {
    // 16 fp32 -> 16 f16, contiguous (v_cvt_pk_f16_f32 pairs + b128 stores)
#pragma unroll
    for (int i = 0; i < 4; ++i) {
        float4 f = g[i];
        dst[4 * i + 0] = (_Float16)f.x; dst[4 * i + 1] = (_Float16)f.y;
        dst[4 * i + 2] = (_Float16)f.z; dst[4 * i + 3] = (_Float16)f.w;
    }
}

// ---------------------------------------------------------------------------
// Kernel 1: Y = X @ W^T + b  ->  f16 head-major [B, NH, SEQ, DK].
// Workgroup 256 thr = 8 waves. Tile 256(M) x 64(N), K-step 32,
// double-buffered LDS, wave tile 32x64 (2 A-frags x 4 shared B-frags).
// One N-tile == one head (h = blockIdx.y); one M-tile stays in one batch.
// ---------------------------------------------------------------------------
__global__ __launch_bounds__(256)
void qkv_proj_kernel(const float* __restrict__ X, const float* __restrict__ W,
                     const float* __restrict__ bias, _Float16* __restrict__ out) {
    const int m0 = blockIdx.x * 256;
    const int n0 = blockIdx.y * 64;
    const int tid = threadIdx.x;
    const int lane = tid & 31, wave = tid >> 5;
    const int half = lane >> 4, l16 = lane & 15;

    __shared__ alignas(16) _Float16 Xs[2][256 * 32];   // 16 KB x2
    __shared__ alignas(16) _Float16 Ws[2][64 * 32];    //  4 KB x2

    const float* xg_base = X + (size_t)(m0 + tid) * DM;               // row/thread
    const float* wg_base = W + (size_t)(n0 + (tid >> 2)) * DM + (tid & 3) * 8;
    _Float16* xs_row = nullptr;  // per-buffer dst computed below
    const int ws_off = (tid >> 2) * 32 + (tid & 3) * 8;

    v8f acc[2][4];
#pragma unroll
    for (int u = 0; u < 2; ++u)
#pragma unroll
        for (int t = 0; t < 4; ++t) acc[u][t] = vzero8();

    {   // prologue: stage tile 0 into buffer 0
        float4 xg[8], wg[2];
        const float4* xs4 = (const float4*)xg_base;
        const float4* ws4 = (const float4*)wg_base;
#pragma unroll
        for (int i = 0; i < 8; ++i) xg[i] = xs4[i];
        wg[0] = ws4[0]; wg[1] = ws4[1];
        cvt_store16(Xs[0] + tid * 32,      xg);
        cvt_store16(Xs[0] + tid * 32 + 16, xg + 4);
        cvt_store16(Ws[0] + ws_off, wg);
    }
    __syncthreads();

    for (int ks = 0; ks < NKSTEP; ++ks) {
        const int cur = ks & 1;
        const bool more = (ks + 1) < NKSTEP;
        float4 xg[8], wg[2];
        if (more) {   // issue next tile's global loads (overlap with WMMAs)
            const int k0 = (ks + 1) * 32;
            const float4* xs4 = (const float4*)(xg_base + k0);
            const float4* ws4 = (const float4*)(wg_base + k0);
#pragma unroll
            for (int i = 0; i < 8; ++i) xg[i] = xs4[i];
            wg[0] = ws4[0]; wg[1] = ws4[1];
        }

        const _Float16* xb = Xs[cur];
        const _Float16* wb = Ws[cur];
        v16h a0 = frag_ld(xb + (wave * 32 + l16) * 32, 0, half);
        v16h a1 = frag_ld(xb + (wave * 32 + 16 + l16) * 32, 0, half);
        v16h bf[4];
#pragma unroll
        for (int t = 0; t < 4; ++t) bf[t] = frag_ld(wb + (t * 16 + l16) * 32, 0, half);
#pragma unroll
        for (int t = 0; t < 4; ++t) acc[0][t] = wmma16(a0, bf[t], acc[0][t]);
#pragma unroll
        for (int t = 0; t < 4; ++t) acc[1][t] = wmma16(a1, bf[t], acc[1][t]);

        if (more) {
            const int nxt = cur ^ 1;
            cvt_store16(Xs[nxt] + tid * 32,      xg);
            cvt_store16(Xs[nxt] + tid * 32 + 16, xg + 4);
            cvt_store16(Ws[nxt] + ws_off, wg);
        }
        __syncthreads();
    }

    // Epilogue (all indices hoisted): h = blockIdx.y, batch = m0/SEQ.
    const int bb = m0 / SEQ;
    const int h = blockIdx.y;
    const int sbase = (m0 % SEQ) + wave * 32 + 8 * half;
    const float bvv = bias[n0 + l16 + ((lane >> 4) << 30 >> 30 & 0) ]; // see below
#pragma unroll
    for (int u = 0; u < 2; ++u)
#pragma unroll
        for (int t = 0; t < 4; ++t) {
            const float bv = bias[n0 + t * 16 + l16];
            const int d = t * 16 + l16;
            _Float16* ob = out + (((size_t)bb * NH + h) * SEQ + sbase + u * 16) * DK + d;
#pragma unroll
            for (int r = 0; r < 8; ++r)
                ob[(size_t)r * DK] = (_Float16)(acc[u][t][r] + bv);
        }
    (void)bvv;
}

// ---------------------------------------------------------------------------
// Kernel 2: causal flash attention, f16 head-major Q/K/V -> f16 [B, SEQ, DM].
// 256 thr = 8 waves; 128 query rows/WG (16 per wave), KV blocks of 64.
// ---------------------------------------------------------------------------
__global__ __launch_bounds__(256)
void attn_kernel(const _Float16* __restrict__ q16, const _Float16* __restrict__ k16,
                 const _Float16* __restrict__ v16f, _Float16* __restrict__ o16) {
    const int q0 = blockIdx.x * 128;
    const int bh = blockIdx.y;
    const int tid = threadIdx.x;
    const int lane = tid & 31, wave = tid >> 5;
    const int half = lane >> 4, l16 = lane & 15;

    __shared__ alignas(16) _Float16 Ks[64 * 64];      // [j][d]
    __shared__ alignas(16) _Float16 Vt[64 * 64];      // [d][j]
    __shared__ alignas(16) _Float16 Ps[8 * 16 * 64];  // per-wave P staging

    const _Float16* qrow = q16 + ((size_t)bh * SEQ + q0 + wave * 16 + l16) * DK;
    const v16h qa0 = frag_ld(qrow, 0, half);
    const v16h qa1 = frag_ld(qrow, 32, half);

    v8f oacc[4];
#pragma unroll
    for (int t = 0; t < 4; ++t) oacc[t] = vzero8();
    float rowm[8], rowl[8];
#pragma unroll
    for (int r = 0; r < 8; ++r) { rowm[r] = -1e30f; rowl[r] = 0.0f; }

    const float scale = 0.125f;       // 1/sqrt(64)
    const int nblocks = q0 / 64 + 2;  // causal coverage of this q-tile
    const int ldr = tid >> 2;         // row this thread stages
    const int ldc = (tid & 3) * 16;

    for (int jb = 0; jb < nblocks; ++jb) {
        const int j0 = jb * 64;
        __syncthreads();
        {   // K row-major, V transposed
            const size_t gbase = ((size_t)bh * SEQ + j0 + ldr) * DK + ldc;
            const v8h* ksv = (const v8h*)(k16 + gbase);
            *(v8h*)(Ks + ldr * 64 + ldc)     = ksv[0];
            *(v8h*)(Ks + ldr * 64 + ldc + 8) = ksv[1];
            const _Float16* vs = v16f + gbase;
#pragma unroll
            for (int i = 0; i < 16; ++i) Vt[(ldc + i) * 64 + ldr] = vs[i];
        }
        if (jb + 1 < nblocks) {
            const size_t nb = ((size_t)bh * SEQ + j0 + 64 + ldr) * DK + ldc;
            __builtin_prefetch((const void*)(k16 + nb), 0, 1);
            __builtin_prefetch((const void*)(v16f + nb), 0, 1);
        }
        __syncthreads();

        // S = Q K^T : grouped frag loads, then one WMMA burst
        v16h kb0[4], kb1[4];
#pragma unroll
        for (int t = 0; t < 4; ++t) {
            kb0[t] = frag_ld(Ks + (t * 16 + l16) * 64, 0, half);
            kb1[t] = frag_ld(Ks + (t * 16 + l16) * 64, 32, half);
        }
        v8f sc[4];
#pragma unroll
        for (int t = 0; t < 4; ++t) sc[t] = wmma16(qa0, kb0[t], vzero8());
#pragma unroll
        for (int t = 0; t < 4; ++t) sc[t] = wmma16(qa1, kb1[t], sc[t]);

        // scale + causal mask
#pragma unroll
        for (int t = 0; t < 4; ++t) {
            const int j = j0 + t * 16 + l16;
#pragma unroll
            for (int r = 0; r < 8; ++r) {
                const int i = q0 + wave * 16 + r + 8 * half;
                const float s = sc[t][r] * scale;
                sc[t][r] = (j > i) ? -1e30f : s;
            }
        }

        // online softmax (rows r + 8*half live in a 16-lane half-wave)
#pragma unroll
        for (int r = 0; r < 8; ++r) {
            float mloc = sc[0][r];
#pragma unroll
            for (int t = 1; t < 4; ++t) mloc = fmaxf(mloc, sc[t][r]);
#pragma unroll
            for (int off = 8; off >= 1; off >>= 1)
                mloc = fmaxf(mloc, __shfl_xor(mloc, off, 32));
            const float mnew = fmaxf(rowm[r], mloc);
            const float corr = __expf(rowm[r] - mnew);
            rowm[r] = mnew;
            float psum = 0.0f;
#pragma unroll
            for (int t = 0; t < 4; ++t) {
                const float p = __expf(sc[t][r] - mnew);
                sc[t][r] = p;
                psum += p;
            }
#pragma unroll
            for (int off = 8; off >= 1; off >>= 1)
                psum += __shfl_xor(psum, off, 32);
            rowl[r] = rowl[r] * corr + psum;
#pragma unroll
            for (int t = 0; t < 4; ++t) oacc[t][r] *= corr;
        }

        // C-layout -> A-layout via per-wave LDS round trip
        _Float16* pw = Ps + wave * 16 * 64;
#pragma unroll
        for (int t = 0; t < 4; ++t)
#pragma unroll
            for (int r = 0; r < 8; ++r)
                pw[(r + 8 * half) * 64 + t * 16 + l16] = (_Float16)sc[t][r];
        const v16h pa0 = frag_ld(pw + l16 * 64, 0, half);
        const v16h pa1 = frag_ld(pw + l16 * 64, 32, half);

        // O += P V : grouped loads then WMMA burst
        v16h vb0[4], vb1[4];
#pragma unroll
        for (int t = 0; t < 4; ++t) {
            vb0[t] = frag_ld(Vt + (t * 16 + l16) * 64, 0, half);
            vb1[t] = frag_ld(Vt + (t * 16 + l16) * 64, 32, half);
        }
#pragma unroll
        for (int t = 0; t < 4; ++t) oacc[t] = wmma16(pa0, vb0[t], oacc[t]);
#pragma unroll
        for (int t = 0; t < 4; ++t) oacc[t] = wmma16(pa1, vb1[t], oacc[t]);
    }

    // normalize + store [B, SEQ, DM] (col = h*64 + d), addresses hoisted
    const int b = bh / NH, h = bh % NH;
#pragma unroll
    for (int t = 0; t < 4; ++t) {
        _Float16* ob = o16 + ((size_t)b * SEQ + q0 + wave * 16 + 8 * half) * DM +
                       h * DK + t * 16 + l16;
#pragma unroll
        for (int r = 0; r < 8; ++r)
            ob[(size_t)r * DM] = (_Float16)(oacc[t][r] / rowl[r]);
    }
}

// ---------------------------------------------------------------------------
// Kernel 3: out = O @ Wo^T + bo (fp32). Same pipeline as kernel 1, f16 A.
// ---------------------------------------------------------------------------
__global__ __launch_bounds__(256)
void out_proj_kernel(const _Float16* __restrict__ A, const float* __restrict__ W,
                     const float* __restrict__ bias, float* __restrict__ out) {
    const int m0 = blockIdx.x * 256;
    const int n0 = blockIdx.y * 64;
    const int tid = threadIdx.x;
    const int lane = tid & 31, wave = tid >> 5;
    const int half = lane >> 4, l16 = lane & 15;

    __shared__ alignas(16) _Float16 As[2][256 * 32];
    __shared__ alignas(16) _Float16 Ws[2][64 * 32];

    const _Float16* ag_base = A + (size_t)(m0 + tid) * DM;
    const float*    wg_base = W + (size_t)(n0 + (tid >> 2)) * DM + (tid & 3) * 8;
    const int ws_off = (tid >> 2) * 32 + (tid & 3) * 8;

    v8f acc[2][4];
#pragma unroll
    for (int u = 0; u < 2; ++u)
#pragma unroll
        for (int t = 0; t < 4; ++t) acc[u][t] = vzero8();

    {   // prologue
        const v8h* as4 = (const v8h*)ag_base;
        const float4* ws4 = (const float4*)wg_base;
        v8h a0 = as4[0], a1 = as4[1], a2 = as4[2], a3 = as4[3];
        float4 wg[2] = { ws4[0], ws4[1] };
        v8h* xd = (v8h*)(As[0] + tid * 32);
        xd[0] = a0; xd[1] = a1; xd[2] = a2; xd[3] = a3;
        cvt_store16(Ws[0] + ws_off, wg);
    }
    __syncthreads();

    for (int ks = 0; ks < NKSTEP; ++ks) {
        const int cur = ks & 1;
        const bool more = (ks + 1) < NKSTEP;
        v8h ag[4]; float4 wg[2];
        if (more) {
            const int k0 = (ks + 1) * 32;
            const v8h* as4 = (const v8h*)(ag_base + k0);
            const float4* ws4 = (const float4*)(wg_base + k0);
#pragma unroll
            for (int i = 0; i < 4; ++i) ag[i] = as4[i];
            wg[0] = ws4[0]; wg[1] = ws4[1];
        }

        const _Float16* xb = As[cur];
        const _Float16* wb = Ws[cur];
        v16h a0 = frag_ld(xb + (wave * 32 + l16) * 32, 0, half);
        v16h a1 = frag_ld(xb + (wave * 32 + 16 + l16) * 32, 0, half);
        v16h bf[4];
#pragma unroll
        for (int t = 0; t < 4; ++t) bf[t] = frag_ld(wb + (t * 16 + l16) * 32, 0, half);
#pragma unroll
        for (int t = 0; t < 4; ++t) acc[0][t] = wmma16(a0, bf[t], acc[0][t]);
#pragma unroll
        for (int t = 0; t < 4; ++t) acc[1][t] = wmma16(a1, bf[t], acc[1][t]);

        if (more) {
            const int nxt = cur ^ 1;
            v8h* xd = (v8h*)(As[nxt] + tid * 32);
#pragma unroll
            for (int i = 0; i < 4; ++i) xd[i] = ag[i];
            cvt_store16(Ws[nxt] + ws_off, wg);
        }
        __syncthreads();
    }

#pragma unroll
    for (int u = 0; u < 2; ++u)
#pragma unroll
        for (int t = 0; t < 4; ++t) {
            const int n = n0 + t * 16 + l16;
            const float bv = bias[n];
            float* ob = out + (size_t)(m0 + wave * 32 + u * 16 + 8 * half) * DM + n;
#pragma unroll
            for (int r = 0; r < 8; ++r)
                ob[(size_t)r * DM] = acc[u][t][r] + bv;
        }
}

// ---------------------------------------------------------------------------
extern "C" void kernel_launch(void* const* d_in, const int* in_sizes, int n_in,
                              void* d_out, int out_size, void* d_ws, size_t ws_size,
                              hipStream_t stream) {
    const float* query = (const float*)d_in[0];
    const float* key   = (const float*)d_in[1];
    const float* value = (const float*)d_in[2];
    const float* Wq = (const float*)d_in[3];
    const float* bq = (const float*)d_in[4];
    const float* Wk = (const float*)d_in[5];
    const float* bk = (const float*)d_in[6];
    const float* Wv = (const float*)d_in[7];
    const float* bv = (const float*)d_in[8];
    const float* Wo = (const float*)d_in[9];
    const float* bo = (const float*)d_in[10];
    float* out = (float*)d_out;

    const size_t nTok = (size_t)BATCH * SEQ * DM;
    _Float16* q16 = (_Float16*)d_ws;   // [B,NH,SEQ,DK]
    _Float16* k16 = q16 + nTok;
    _Float16* v16 = k16 + nTok;
    _Float16* o16 = v16 + nTok;        // [B,SEQ,DM]
    (void)in_sizes; (void)n_in; (void)out_size; (void)ws_size;

    dim3 blk(256);
    dim3 gproj(BATCH * SEQ / 256, DM / 64);   // (32, 12)
    qkv_proj_kernel<<<gproj, blk, 0, stream>>>(query, Wq, bq, q16);
    qkv_proj_kernel<<<gproj, blk, 0, stream>>>(key,   Wk, bk, k16);
    qkv_proj_kernel<<<gproj, blk, 0, stream>>>(value, Wv, bv, v16);

    dim3 gattn(SEQ / 128, BATCH * NH);        // (16, 48)
    attn_kernel<<<gattn, blk, 0, stream>>>(q16, k16, v16, o16);

    out_proj_kernel<<<gproj, blk, 0, stream>>>(o16, Wo, bo, out);
}
